// MutiHeadAttention_9878424781404
// MI455X (gfx1250) — matrix operand order repaired and checked by reading, not compiled, not run on hardware
//
#include <hip/hip_runtime.h>

// Problem constants (from the reference)
#define Bz  2
#define Sz  2048
#define Dz  1024      // D_IN == D_OUT
#define Hz  16
#define HDz 64
#define Mz  (Bz * Sz) // 4096 rows

typedef __attribute__((ext_vector_type(16))) __bf16          v16bf;
typedef __attribute__((ext_vector_type(8)))  float           v8f;
typedef __attribute__((ext_vector_type(8)))  unsigned short  us8;
typedef __attribute__((ext_vector_type(16))) unsigned short  us16;

union FragBF { us16 u; v16bf b; };

__device__ __forceinline__ unsigned short f2bf(float f) {
  unsigned int u = __float_as_uint(f);
  u += 0x7FFFu + ((u >> 16) & 1u);          // round-to-nearest-even
  return (unsigned short)(u >> 16);
}

__device__ __forceinline__ v8f vzero8() {
  v8f z = {0.f, 0.f, 0.f, 0.f, 0.f, 0.f, 0.f, 0.f};
  return z;
}

// Build a v16bf fragment from two contiguous 16-byte chunks.
__device__ __forceinline__ FragBF frag_from(const unsigned short* p0,
                                            const unsigned short* p1) {
  us8 lo = *(const us8*)p0;
  us8 hi = *(const us8*)p1;
  FragBF f;
  f.u = __builtin_shufflevector(lo, hi, 0,1,2,3,4,5,6,7,8,9,10,11,12,13,14,15);
  return f;
}

__device__ __forceinline__ v8f wmma_bf16(const FragBF& a, const FragBF& b, v8f c) {
  return __builtin_amdgcn_wmma_f32_16x16x32_bf16(
      /*neg_a=*/false, a.b, /*neg_b=*/false, b.b,
      /*c_mod=*/(short)0, c, /*reuse_a=*/false, /*reuse_b=*/false);
}

// ---------------------------------------------------------------- prep passes
__global__ void k_f32_to_bf16(const float* __restrict__ in,
                              unsigned short* __restrict__ out, int n) {
  int i = blockIdx.x * blockDim.x + threadIdx.x;
  if (i < n) out[i] = f2bf(in[i]);
}

// W [K,N] fp32 -> WT [N,K] bf16 (so WMMA B-operand loads are contiguous)
__global__ void k_transpose_bf16(const float* __restrict__ W,
                                 unsigned short* __restrict__ WT, int K, int N) {
  int i = blockIdx.x * blockDim.x + threadIdx.x;
  if (i >= K * N) return;
  int k = i / N, n = i % N;
  WT[(size_t)n * K + k] = f2bf(W[i]);
}

// ---------------------------------------------------------------- WMMA GEMM
// C[M,N] = A[M,K] (bf16 row-major) * BT[N,K]^T (bf16, pre-transposed)
// One wave (32 threads) computes a 64x64 tile: 4x4 grid of 16x16 WMMA tiles.
// MODE 0: store bf16 row-major [M,N]
// MODE 1: store bf16 per-head-transposed V^T: [B,H,HD,S]
// MODE 2: store f32 row-major [M,N] + bias[n]
// MODE 3: store bf16 row-major [M,N], pre-scaled by 0.125 (Q / sqrt(HD))
template <int MODE>
__global__ __launch_bounds__(32)
void k_gemm_wmma(const unsigned short* __restrict__ A,
                 const unsigned short* __restrict__ BT,
                 unsigned short* __restrict__ outB,
                 float* __restrict__ outF,
                 const float* __restrict__ bias,
                 int M, int N, int K) {
  const int lane = threadIdx.x;
  const int col  = lane & 15;   // N (or M-row for A frags)
  const int half = lane >> 4;   // which K-half this lane holds
  const int m0 = blockIdx.x * 64;
  const int n0 = blockIdx.y * 64;

  v8f acc[4][4];
#pragma unroll
  for (int mi = 0; mi < 4; mi++)
#pragma unroll
    for (int ni = 0; ni < 4; ni++) acc[mi][ni] = vzero8();

  for (int k0 = 0; k0 < K; k0 += 32) {
    FragBF af[4], bf[4];
#pragma unroll
    for (int mi = 0; mi < 4; mi++) {
      // A 16x32 frag: lane holds row m0+mi*16+col, K = half*16 .. +15 contiguous
      const unsigned short* ap =
          A + (size_t)(m0 + mi * 16 + col) * K + k0 + half * 16;
      af[mi] = frag_from(ap, ap + 8);
    }
#pragma unroll
    for (int ni = 0; ni < 4; ni++) {
      // B 32x16 frag: lane holds col n0+ni*16+col, K = half*16 .. +15 contiguous
      const unsigned short* bp =
          BT + (size_t)(n0 + ni * 16 + col) * K + k0 + half * 16;
      bf[ni] = frag_from(bp, bp + 8);
    }
#pragma unroll
    for (int mi = 0; mi < 4; mi++)
#pragma unroll
      for (int ni = 0; ni < 4; ni++)
        acc[mi][ni] = wmma_bf16(af[mi], bf[ni], acc[mi][ni]);
  }

  // Epilogue. C-layout: elem g -> row m0+mi*16+g+half*8, col n0+ni*16+col
#pragma unroll
  for (int mi = 0; mi < 4; mi++) {
#pragma unroll
    for (int ni = 0; ni < 4; ni++) {
#pragma unroll
      for (int g = 0; g < 8; g++) {
        int m = m0 + mi * 16 + g + half * 8;
        int n = n0 + ni * 16 + col;
        float v = acc[mi][ni][g];
        if constexpr (MODE == 0) {
          outB[(size_t)m * N + n] = f2bf(v);
        } else if constexpr (MODE == 1) {
          int b = m / Sz, s = m % Sz;
          int h = n / HDz, d = n % HDz;
          outB[(((size_t)b * Hz + h) * HDz + d) * Sz + s] = f2bf(v);
        } else if constexpr (MODE == 2) {
          outF[(size_t)m * N + n] = v + bias[n];
        } else {
          outB[(size_t)m * N + n] = f2bf(v * 0.125f);  // fold 1/sqrt(HD)
        }
      }
    }
  }
}

// ---------------------------------------------------------------- flash attn
// Process one 32-key block against a 16-query tile (online softmax).
// MASK=false: block is strictly below the diagonal -> no causal compares.
template <bool MASK>
__device__ __forceinline__ void attn_block(
    int kb, int col, int half, int qrowBase,
    const unsigned short* __restrict__ Kp,
    const unsigned short* __restrict__ Vp,
    const FragBF (&qf)[2], v8f (&acc)[4],
    float (&runMax)[8], float (&runSum)[8],
    unsigned short* Pld) {
  // scores: two 16x16 tiles (keys kb*32..+15 and +16..+31); Q pre-scaled
  v8f s0 = vzero8(), s1 = vzero8();
#pragma unroll
  for (int dk = 0; dk < 2; dk++) {
    const unsigned short* kp0 =
        Kp + (size_t)(kb * 32 + col) * Dz + dk * 32 + half * 16;
    const unsigned short* kp1 =
        Kp + (size_t)(kb * 32 + 16 + col) * Dz + dk * 32 + half * 16;
    FragBF kf0 = frag_from(kp0, kp0 + 8);
    FragBF kf1 = frag_from(kp1, kp1 + 8);
    s0 = wmma_bf16(qf[dk], kf0, s0);
    s1 = wmma_bf16(qf[dk], kf1, s1);
  }

  const int kg0 = kb * 32 + col;
  const int kg1 = kg0 + 16;
#pragma unroll
  for (int g = 0; g < 8; g++) {
    float a0 = s0[g];
    float a1 = s1[g];
    if constexpr (MASK) {
      const int qg = qrowBase + g;
      if (kg0 > qg) a0 = -1e30f;
      if (kg1 > qg) a1 = -1e30f;
    }
    float m = fmaxf(a0, a1);
    m = fmaxf(m, __shfl_xor(m, 1, 32));
    m = fmaxf(m, __shfl_xor(m, 2, 32));
    m = fmaxf(m, __shfl_xor(m, 4, 32));
    m = fmaxf(m, __shfl_xor(m, 8, 32));
    float nm   = fmaxf(runMax[g], m);
    float corr = __expf(runMax[g] - nm);
    float p0 = __expf(a0 - nm);
    float p1 = __expf(a1 - nm);
    float ps = p0 + p1;
    ps += __shfl_xor(ps, 1, 32);
    ps += __shfl_xor(ps, 2, 32);
    ps += __shfl_xor(ps, 4, 32);
    ps += __shfl_xor(ps, 8, 32);
    runSum[g] = runSum[g] * corr + ps;
    runMax[g] = nm;
#pragma unroll
    for (int ni = 0; ni < 4; ni++) acc[ni][g] *= corr;
    // stash P (C-layout -> LDS row-major 16x32)
    const int row = half * 8 + g;
    Pld[row * 32 + col]      = f2bf(p0);
    Pld[row * 32 + 16 + col] = f2bf(p1);
  }
  __syncthreads();  // single-wave WG: barrier is NOP, keeps DS ordering

  // reload P in A-fragment layout (16 queries x 32 keys)
  const unsigned short* pp = &Pld[col * 32 + half * 16];
  FragBF pf = frag_from(pp, pp + 8);
  __syncthreads();  // protect Pld before next block overwrites it

  // ctx += P * V  (B operand: Vt rows are contiguous over keys)
#pragma unroll
  for (int ni = 0; ni < 4; ni++) {
    const unsigned short* vp =
        Vp + (size_t)(ni * 16 + col) * Sz + kb * 32 + half * 16;
    FragBF vf = frag_from(vp, vp + 8);
    acc[ni] = wmma_bf16(pf, vf, acc[ni]);
  }
}

// One wave per (b, h, 16-query tile). Q,K: bf16 row-major [B*S, D] (head
// offset h*HD, Q pre-scaled by 1/sqrt(HD)). Vt: bf16 [B,H,HD,S].
// ctx out: bf16 row-major [B*S, D].
__global__ __launch_bounds__(32)
void k_flash_attn(const unsigned short* __restrict__ Q,
                  const unsigned short* __restrict__ Km,
                  const unsigned short* __restrict__ Vt,
                  unsigned short* __restrict__ ctx) {
  __shared__ __align__(16) unsigned short Pld[16 * 32];  // P tile, row-major

  const int lane = threadIdx.x;
  const int col  = lane & 15;
  const int half = lane >> 4;
  const int qt = blockIdx.x;   // query tile
  const int h  = blockIdx.y;
  const int b  = blockIdx.z;

  const unsigned short* Qp = Q  + (size_t)b * Sz * Dz + (size_t)h * HDz;
  const unsigned short* Kp = Km + (size_t)b * Sz * Dz + (size_t)h * HDz;
  const unsigned short* Vp = Vt + ((size_t)b * Hz + h) * HDz * Sz; // [HD][S]

  // Q fragments: 16 queries x 64 d, split into two K=32 chunks
  FragBF qf[2];
#pragma unroll
  for (int dk = 0; dk < 2; dk++) {
    const unsigned short* qp =
        Qp + (size_t)(qt * 16 + col) * Dz + dk * 32 + half * 16;
    qf[dk] = frag_from(qp, qp + 8);
  }

  v8f acc[4];              // ctx 16x64 accumulator (4 d-tiles, C-layout)
#pragma unroll
  for (int ni = 0; ni < 4; ni++) acc[ni] = vzero8();
  float runMax[8], runSum[8];
#pragma unroll
  for (int g = 0; g < 8; g++) { runMax[g] = -1e30f; runSum[g] = 0.f; }

  const int qrowBase = qt * 16 + half * 8;
  const int nkb = (qt * 16 + 16 + 31) / 32;  // causal key-block bound
  // All blocks except the last are strictly below the diagonal: a 16-query
  // tile never straddles two 32-key blocks on the diagonal.
  for (int kb = 0; kb < nkb - 1; kb++)
    attn_block<false>(kb, col, half, qrowBase, Kp, Vp, qf, acc,
                      runMax, runSum, Pld);
  attn_block<true>(nkb - 1, col, half, qrowBase, Kp, Vp, qf, acc,
                   runMax, runSum, Pld);

  // normalize and store ctx as bf16 row-major [B*S, D]
#pragma unroll
  for (int g = 0; g < 8; g++) {
    const float inv = 1.0f / runSum[g];
    const int row = qt * 16 + half * 8 + g;
#pragma unroll
    for (int ni = 0; ni < 4; ni++) {
      float v = acc[ni][g] * inv;
      ctx[((size_t)b * Sz + row) * Dz + h * HDz + ni * 16 + col] = f2bf(v);
    }
  }
}

// ---------------------------------------------------------------- launcher
extern "C" void kernel_launch(void* const* d_in, const int* in_sizes, int n_in,
                              void* d_out, int out_size, void* d_ws, size_t ws_size,
                              hipStream_t stream) {
  (void)in_sizes; (void)n_in; (void)out_size; (void)ws_size;
  const float* x  = (const float*)d_in[0];
  const float* Wq = (const float*)d_in[1];
  const float* Wk = (const float*)d_in[2];
  const float* Wv = (const float*)d_in[3];
  const float* Wo = (const float*)d_in[4];
  const float* bo = (const float*)d_in[5];
  float* out = (float*)d_out;

  unsigned short* ws = (unsigned short*)d_ws;
  size_t off = 0;
  unsigned short* xb  = ws + off; off += (size_t)Mz * Dz;   // x  bf16
  unsigned short* WqT = ws + off; off += (size_t)Dz * Dz;   // Wq^T bf16
  unsigned short* WkT = ws + off; off += (size_t)Dz * Dz;
  unsigned short* WvT = ws + off; off += (size_t)Dz * Dz;
  unsigned short* WoT = ws + off; off += (size_t)Dz * Dz;
  unsigned short* Qb  = ws + off; off += (size_t)Mz * Dz;   // Q bf16 [M,D] (scaled)
  unsigned short* Kb  = ws + off; off += (size_t)Mz * Dz;   // K bf16 [M,D]
  unsigned short* Vtb = ws + off; off += (size_t)Mz * Dz;   // V^T bf16 [B,H,HD,S]
  unsigned short* Ctx = ws + off; off += (size_t)Mz * Dz;   // ctx bf16 [M,D]

  const int nX = Mz * Dz;       // 4,194,304
  const int nW = Dz * Dz;       // 1,048,576
  k_f32_to_bf16<<<nX / 256, 256, 0, stream>>>(x, xb, nX);
  k_transpose_bf16<<<nW / 256, 256, 0, stream>>>(Wq, WqT, Dz, Dz);
  k_transpose_bf16<<<nW / 256, 256, 0, stream>>>(Wk, WkT, Dz, Dz);
  k_transpose_bf16<<<nW / 256, 256, 0, stream>>>(Wv, WvT, Dz, Dz);
  k_transpose_bf16<<<nW / 256, 256, 0, stream>>>(Wo, WoT, Dz, Dz);

  dim3 ggrid(Mz / 64, Dz / 64);  // 64 x 16 waves
  k_gemm_wmma<3><<<ggrid, 32, 0, stream>>>(xb, WqT, Qb,  nullptr, nullptr, Mz, Dz, Dz);
  k_gemm_wmma<0><<<ggrid, 32, 0, stream>>>(xb, WkT, Kb,  nullptr, nullptr, Mz, Dz, Dz);
  k_gemm_wmma<1><<<ggrid, 32, 0, stream>>>(xb, WvT, Vtb, nullptr, nullptr, Mz, Dz, Dz);

  dim3 agrid(Sz / 16, Hz, Bz);   // 128 x 16 x 2 waves
  k_flash_attn<<<agrid, 32, 0, stream>>>(Qb, Kb, Vtb, Ctx);

  k_gemm_wmma<2><<<ggrid, 32, 0, stream>>>(Ctx, WoT, nullptr, out, bo, Mz, Dz, Dz);
}
